// RBFKernel_42958262894792
// MI455X (gfx1250) — matrix operand order repaired and checked by reading, not compiled, not run on hardware
//
#include <hip/hip_runtime.h>

// ---------------------------------------------------------------------------
// RBF kernel for MI455X (gfx1250): exp(-||x_i - x1_j||^2 / h), h = 1.
//
// Pipeline:
//   1) rbf_rownorms : ||x_i||^2, ||x1_j||^2 -> ws
//   2) rbf_split    : f32 -> bf16 hi/lo (x = hi + lo), stored in K-panel
//                     layout [K/32][8192][32] so every 128x32 tile is one
//                     contiguous 8KB block
//   3) rbf_wmma     : 8192x8192x512 GEMM as bf16x3 (hi*hi + hi*lo + lo*hi)
//                     on v_wmma_f32_16x16x32_bf16, double-buffered LDS fed by
//                     GLOBAL_LOAD_ASYNC_TO_LDS_B128 (ASYNCcnt), fused
//                     norm + exp epilogue.
//
// Workspace layout (assumes ws_size >= 32 MB + 64 KB):
//   [0, 64KB)        f32 norms: x (8192) then x1 (8192)
//   [64KB, +8MB)     Xhi panels    (bf16)
//   [+8MB, +16MB)    Xlo panels
//   [+16MB, +24MB)   X1hi panels
//   [+24MB, +32MB)   X1lo panels
// ---------------------------------------------------------------------------

typedef __attribute__((ext_vector_type(16))) __bf16 v16bf;
typedef __attribute__((ext_vector_type(8)))  float  v8f;

#define NROWS 8192
#define DK    512
#define BM    128
#define BN    128
#define BK    32
#define LDSTR       40                      // halves per LDS row (32 + 8 pad) -> 80B
#define TILE_HALVES (BM * LDSTR)            // 5120 halves = 10240 B per array
#define PANEL_ELEMS ((size_t)NROWS * DK)    // halves per hi/lo matrix

// Round-to-nearest-even f32 -> bf16 (integer path)
__device__ __forceinline__ unsigned short bf16_rne(float f) {
    unsigned int u = __float_as_uint(f);
    u += 0x7FFFu + ((u >> 16) & 1u);
    return (unsigned short)(u >> 16);
}

// Split f32 into bf16 hi (truncated -> exact residual) + bf16 lo (RNE)
__device__ __forceinline__ void split_bf16(float x, unsigned short& hi, unsigned short& lo) {
    unsigned int u  = __float_as_uint(x);
    unsigned int uh = u & 0xFFFF0000u;
    hi = (unsigned short)(uh >> 16);
    lo = bf16_rne(x - __uint_as_float(uh));
}

// A fragment (16x32 bf16, MxK): lane<16 -> K {hf*8..+7} U {16+hf*8..+7}
__device__ __forceinline__ v16bf frag_a(const unsigned short* p, int hf) {
    struct U { uint4 a, b; } u;
    u.a = *(const uint4*)(p + hf * 8);
    u.b = *(const uint4*)(p + 16 + hf * 8);
    return __builtin_bit_cast(v16bf, u);
}

// B fragment (32x16 bf16, KxN stored as N rows of K): lane<16 -> K {hf*8..+7} U {16+hf*8..+7}
__device__ __forceinline__ v16bf frag_b(const unsigned short* p, int hf) {
    struct U { uint4 a, b; } u;
    u.a = *(const uint4*)(p + hf * 8);
    u.b = *(const uint4*)(p + 16 + hf * 8);
    return __builtin_bit_cast(v16bf, u);
}

// ---------------------------------------------------------------------------
// Row-norm precompute
// ---------------------------------------------------------------------------
__global__ void rbf_rownorms(const float* __restrict__ X,
                             const float* __restrict__ X1,
                             float* __restrict__ ws) {
    const float* src = blockIdx.y ? X1 : X;
    float* dst = ws + (size_t)blockIdx.y * NROWS;
    const int row = blockIdx.x;
    const int t   = threadIdx.x;               // 128 threads, one float4 each
    float4 v = *(const float4*)(src + (size_t)row * DK + t * 4);
    float s = v.x * v.x + v.y * v.y + v.z * v.z + v.w * v.w;
    __shared__ float red[128];
    red[t] = s;
    __syncthreads();
    for (int o = 64; o > 0; o >>= 1) {
        if (t < o) red[t] += red[t + o];
        __syncthreads();
    }
    if (t == 0) dst[row] = red[0];
}

// ---------------------------------------------------------------------------
// One-shot bf16 hi/lo split into K-panel layout [DK/32][NROWS][32]
// ---------------------------------------------------------------------------
__global__ __launch_bounds__(256)
void rbf_split(const float* __restrict__ X, const float* __restrict__ X1,
               unsigned short* __restrict__ ws16) {
    const int mat = blockIdx.y;
    const float* src = mat ? X1 : X;
    unsigned short* hi = ws16 + (size_t)mat * 2 * PANEL_ELEMS;
    unsigned short* lo = hi + PANEL_ELEMS;

    const size_t idx = ((size_t)blockIdx.x * 256 + threadIdx.x) * 4;
    const int row = (int)(idx >> 9);     // / DK
    const int col = (int)(idx & 511);

    float4 v = *(const float4*)(src + idx);
    ushort4 h, l;
    split_bf16(v.x, h.x, l.x);
    split_bf16(v.y, h.y, l.y);
    split_bf16(v.z, h.z, l.z);
    split_bf16(v.w, h.w, l.w);

    const size_t o = ((size_t)(col >> 5) * NROWS + row) * 32 + (col & 31);
    *(ushort4*)(hi + o) = h;
    *(ushort4*)(lo + o) = l;
}

// ---------------------------------------------------------------------------
// Async copy of one K-step tile set (Ahi, Alo, Bhi, Blo; each 128x32 bf16 =
// 8KB contiguous in global) into an LDS buffer via GLOBAL_LOAD_ASYNC_TO_LDS.
// 256 threads x 8 x 16B = 32KB; ASYNCcnt += 8 per wave.
// ---------------------------------------------------------------------------
__device__ __forceinline__ void async_copy_tiles(unsigned ldsbase,
                                                 const unsigned short* g0,
                                                 const unsigned short* g1,
                                                 const unsigned short* g2,
                                                 const unsigned short* g3,
                                                 int tid) {
    const unsigned short* gp[4] = {g0, g1, g2, g3};
#pragma unroll
    for (int a = 0; a < 4; ++a) {
#pragma unroll
        for (int rep = 0; rep < 2; ++rep) {
            const int j = tid + rep * 256;                 // 16B chunk id in tile
            const unsigned lds  = ldsbase + (unsigned)a * 10240u
                                + (unsigned)(j >> 2) * 80u + (unsigned)(j & 3) * 16u;
            const unsigned voff = (unsigned)j * 16u;       // byte offset in panel tile
            asm volatile("global_load_async_to_lds_b128 %0, %1, %2"
                         :: "v"(lds), "v"(voff), "s"(gp[a])
                         : "memory");
        }
    }
}

// ---------------------------------------------------------------------------
// Main fused GEMM + RBF epilogue. 256 threads = 8 waves; 128x128 tile/block.
// Wave grid 4(M) x 2(N): each wave owns 32x64 = 2x4 WMMA tiles of 16x16.
// ---------------------------------------------------------------------------
__global__ __launch_bounds__(256)
void rbf_wmma(const unsigned short* __restrict__ Ahi_g,
              const unsigned short* __restrict__ Alo_g,
              const unsigned short* __restrict__ Bhi_g,
              const unsigned short* __restrict__ Blo_g,
              const float* __restrict__ xsq, const float* __restrict__ x1sq,
              float* __restrict__ out) {
    __shared__ unsigned short LDS[2 * 4 * TILE_HALVES];   // 80 KB, double buffered
    __shared__ float sAsq[BM];
    __shared__ float sBsq[BN];

    const int tid  = threadIdx.x;
    const int lane = tid & 31;
    const int wid  = tid >> 5;
    const int wm   = wid & 3;    // 0..3 -> 32-row strip of M
    const int wn   = wid >> 2;   // 0..1 -> 64-col strip of N

    const int row0 = blockIdx.y * BM;   // rows of X   (output rows)
    const int col0 = blockIdx.x * BN;   // rows of X1  (output cols)

    if (tid < BM)           sAsq[tid]      = xsq[row0 + tid];
    else                    sBsq[tid - BM] = x1sq[col0 + (tid - BM)];

    v8f acc[2][4];
    const v8f vzero = {};
#pragma unroll
    for (int mt = 0; mt < 2; ++mt)
#pragma unroll
        for (int nt = 0; nt < 4; ++nt) acc[mt][nt] = vzero;

    const int hf = lane >> 4;       // K-half selector
    const int mr = lane & 15;       // M (or N) index within a 16-tile

    const unsigned lds0 = (unsigned)(size_t)&LDS[0];

    // Panel tile base for K-chunk kc: (kc * NROWS + rowbase) * 32 halves.
    auto tileA = [&](const unsigned short* g, int kc) {
        return g + ((size_t)kc * NROWS + row0) * 32;
    };
    auto tileB = [&](const unsigned short* g, int kc) {
        return g + ((size_t)kc * NROWS + col0) * 32;
    };

    // Prologue: fill buffer 0 with K-chunk 0.
    async_copy_tiles(lds0, tileA(Ahi_g, 0), tileA(Alo_g, 0),
                           tileB(Bhi_g, 0), tileB(Blo_g, 0), tid);

    int p = 0;
    for (int k0 = 0; k0 < DK; k0 += BK, p ^= 1) {
        __syncthreads();   // everyone finished reading buffer p^1 (prev iter)

        if (k0 + BK < DK) {
            const int kc = (k0 + BK) >> 5;
            async_copy_tiles(lds0 + (unsigned)(p ^ 1) * (4u * 10240u),
                             tileA(Ahi_g, kc), tileA(Alo_g, kc),
                             tileB(Bhi_g, kc), tileB(Blo_g, kc), tid);
            asm volatile("s_wait_asynccnt 8" ::: "memory");   // buffer p done
        } else {
            asm volatile("s_wait_asynccnt 0" ::: "memory");
        }
        __syncthreads();   // all waves' buffer-p loads visible

        const unsigned short* buf = &LDS[p * 4 * TILE_HALVES];

        // -------- LDS -> fragments ----------------------------------------
        v16bf ah[2], al[2], bh[4], bl[4];
#pragma unroll
        for (int mt = 0; mt < 2; ++mt) {
            const int r = (wm * 32 + mt * 16 + mr) * LDSTR;
            ah[mt] = frag_a(buf + 0 * TILE_HALVES + r, hf);
            al[mt] = frag_a(buf + 1 * TILE_HALVES + r, hf);
        }
#pragma unroll
        for (int nt = 0; nt < 4; ++nt) {
            const int r = (wn * 64 + nt * 16 + mr) * LDSTR;
            bh[nt] = frag_b(buf + 2 * TILE_HALVES + r, hf);
            bl[nt] = frag_b(buf + 3 * TILE_HALVES + r, hf);
        }

        // -------- bf16x3 WMMA: hi*hi + hi*lo + lo*hi ----------------------
#pragma unroll
        for (int mt = 0; mt < 2; ++mt)
#pragma unroll
            for (int nt = 0; nt < 4; ++nt) {
                acc[mt][nt] = __builtin_amdgcn_wmma_f32_16x16x32_bf16(
                    false, ah[mt], false, bh[nt], (short)0, acc[mt][nt], false, false);
                acc[mt][nt] = __builtin_amdgcn_wmma_f32_16x16x32_bf16(
                    false, ah[mt], false, bl[nt], (short)0, acc[mt][nt], false, false);
                acc[mt][nt] = __builtin_amdgcn_wmma_f32_16x16x32_bf16(
                    false, al[mt], false, bh[nt], (short)0, acc[mt][nt], false, false);
            }
    }

    // -------- fused epilogue: exp(-max(xsq + x1sq - 2*cross, 0)) ----------
#pragma unroll
    for (int mt = 0; mt < 2; ++mt) {
#pragma unroll
        for (int nt = 0; nt < 4; ++nt) {
            const int mloc = wm * 32 + mt * 16 + hf * 8;  // + v gives the row
            const int nloc = wn * 64 + nt * 16 + mr;
            const float bs = sBsq[nloc];
            float* orow = out + (size_t)(row0 + mloc) * NROWS + (col0 + nloc);
#pragma unroll
            for (int v = 0; v < 8; ++v) {
                float d = sAsq[mloc + v] + bs - 2.0f * acc[mt][nt][v];
                d = fmaxf(d, 0.0f);
                orow[(size_t)v * NROWS] = __expf(-d);
            }
        }
    }
}

// ---------------------------------------------------------------------------
extern "C" void kernel_launch(void* const* d_in, const int* in_sizes, int n_in,
                              void* d_out, int out_size, void* d_ws, size_t ws_size,
                              hipStream_t stream) {
    const float* x  = (const float*)d_in[0];
    const float* x1 = (const float*)d_in[1];
    float* out = (float*)d_out;

    float* norms = (float*)d_ws;                         // 16384 f32
    unsigned short* ws16 = (unsigned short*)(norms + 2 * NROWS);
    unsigned short* Ahi = ws16;                          // [DK/32][NROWS][32]
    unsigned short* Alo = Ahi + PANEL_ELEMS;
    unsigned short* Bhi = Alo + PANEL_ELEMS;
    unsigned short* Blo = Bhi + PANEL_ELEMS;

    dim3 gN(NROWS, 2);
    rbf_rownorms<<<gN, 128, 0, stream>>>(x, x1, norms);

    dim3 gS((unsigned)(PANEL_ELEMS / (256 * 4)), 2);
    rbf_split<<<gS, 256, 0, stream>>>(x, x1, ws16);

    dim3 gG(NROWS / BN, NROWS / BM);
    rbf_wmma<<<gG, 256, 0, stream>>>(Ahi, Alo, Bhi, Blo,
                                     norms, norms + NROWS, out);
}
